// EdgeConvLayer_75024488727174
// MI455X (gfx1250) — matrix operand order repaired and checked by reading, not compiled, not run on hardware
//
#include <hip/hip_runtime.h>

// ---------------------------------------------------------------------------
// EdgeConv on gfx1250 (MI455X): bf16 WMMA GEMM + monotone-BN max trick.
//   B=4, N=16384, C=64, K=16, O=64, 2C=128.
// Pass 1: per (b,n) point, one wave computes h = edge(16x128) @ W^T(128x64)
//         with 16 x v_wmma_f32_16x16x32_bf16; stores max_k h -> d_out,
//         min_k h -> ws, and per-block sum/sumsq partials (deterministic).
// Pass 2: fold partials -> per-channel scale/shift.
// Pass 3: out = relu((scale>=0 ? max : min)*scale + shift) in place.
// ---------------------------------------------------------------------------

typedef __attribute__((ext_vector_type(16))) __bf16 v16bf;
typedef __attribute__((ext_vector_type(8)))  float  v8f;

#define B_    4
#define N_    16384
#define C_    64
#define K_    16
#define O_    64
#define TWOC_ 128
#define EPS_  1e-5f

#define BLOCKS_ 1024
#define PPB_    ((B_ * N_) / BLOCKS_)   // 64 points per block
#define ITERS_  (PPB_ / 8)              // 8 iterations, 8 waves/block

union AFrag { uint4 q[2]; v16bf v; };
struct __align__(8) bf4 { __bf16 h[4]; };

__global__ __launch_bounds__(256)
void edgeconv_gemm_kernel(const float* __restrict__ x,
                          const int*   __restrict__ idx,
                          const float* __restrict__ weight,
                          float* __restrict__ hmax,      // aliases d_out
                          float* __restrict__ hmin,      // ws
                          float* __restrict__ partial)   // ws: [BLOCKS_][128]
{
    // Per-wave LDS staging region: 16 rows x 128 bf16, pitch 136 (272B) to
    // spread the 16 row-major ds_load_b128 A-fragment reads across banks.
    __shared__ __align__(16) __bf16 edgeLds[8][16][136];
    __shared__ float redS[8][64];
    __shared__ float redQ[8][64];

    const int tid  = threadIdx.x;
    const int lane = tid & 31;
    const int wv   = tid >> 5;       // wave in block: 0..7
    const int col  = lane & 15;      // N-column / A-row within fragment
    const int hi   = lane >> 4;      // half-wave select

    // ---- B fragments: W^T in bf16, register resident (16 x v16bf) --------
    // B fragment (32x16, 16-bit), contiguous k order used here: lane holds column N=col, K range hi*16..+15,
    // packed 2 per VGPR. weight is [O][2C] row-major, so the 16 K values for
    // a fixed o are 16 contiguous floats.
    v16bf Bf[4][4];
#pragma unroll
    for (int t = 0; t < 4; ++t) {
#pragma unroll
        for (int kc = 0; kc < 4; ++kc) {
            const int o  = t * 16 + col;
            const int kb = kc * 32 + hi * 16;
            const float* wp = weight + o * TWOC_ + kb;
            union { __bf16 h[16]; v16bf v; } u;
#pragma unroll
            for (int j = 0; j < 16; ++j) u.h[j] = (__bf16)wp[j];
            Bf[t][kc] = u.v;
        }
    }

    const v8f vzero = {0.f, 0.f, 0.f, 0.f, 0.f, 0.f, 0.f, 0.f};
    v8f accS[4], accQ[4];
#pragma unroll
    for (int t = 0; t < 4; ++t) { accS[t] = vzero; accQ[t] = vzero; }

    // Staging lane roles: lane = (k neighbor)*2 + half-of-channels.
    const int kk = lane >> 1;   // 0..15 neighbor index
    const int hf = lane & 1;    // channel half: [0,32) or [32,64)

    for (int it = 0; it < ITERS_; ++it) {
        const int pt = blockIdx.x * PPB_ + it * 8 + wv;   // (b,n) flat
        const int b  = pt >> 14;
        const int n  = pt & (N_ - 1);

        const float* xb = x + (size_t)b * (N_ * C_);
        const int    nb = idx[(size_t)pt * K_ + kk];
        const float4* np4 = (const float4*)(xb + (size_t)nb * C_ + hf * 32);
        const float4* cp4 = (const float4*)(xb + (size_t)n  * C_ + hf * 32);

        __bf16* rowc = &edgeLds[wv][kk][hf * 32];        // center cols
        __bf16* rowd = &edgeLds[wv][kk][64 + hf * 32];   // diff cols
#pragma unroll
        for (int j = 0; j < 8; ++j) {
            const float4 cv = cp4[j];   // center chunk (L0/L2 hit, shared)
            const float4 nv = np4[j];   // gathered neighbor chunk
            bf4 tc, td;
            tc.h[0] = (__bf16)cv.x;        tc.h[1] = (__bf16)cv.y;
            tc.h[2] = (__bf16)cv.z;        tc.h[3] = (__bf16)cv.w;
            td.h[0] = (__bf16)(nv.x - cv.x); td.h[1] = (__bf16)(nv.y - cv.y);
            td.h[2] = (__bf16)(nv.z - cv.z); td.h[3] = (__bf16)(nv.w - cv.w);
            *(bf4*)(rowc + j * 4) = tc;
            *(bf4*)(rowd + j * 4) = td;
        }
        __syncthreads();   // LDS writes visible (and compiler-ordered)

        // ---- 16 WMMAs: D[k][o] += edge[16x32] x W^T[32x16] ---------------
        v8f Cc[4];
#pragma unroll
        for (int t = 0; t < 4; ++t) Cc[t] = vzero;
#pragma unroll
        for (int kc = 0; kc < 4; ++kc) {
            // A fragment (16x32, 16-bit), same contiguous k order as B: lanes 0-15 row=col hold K 0-15;
            // lanes 16-31 hold K 16-31 -> two contiguous 16B LDS reads.
            const __bf16* ap = &edgeLds[wv][col][kc * 32 + hi * 16];
            AFrag a;
            a.q[0] = *(const uint4*)(ap);
            a.q[1] = *(const uint4*)(ap + 8);
#pragma unroll
            for (int t = 0; t < 4; ++t) {
                Cc[t] = __builtin_amdgcn_wmma_f32_16x16x32_bf16(
                    false, a.v, false, Bf[t][kc], (short)0, Cc[t],
                    false, false);
            }
        }
        __syncthreads();   // A reads done before next iteration's writes

        // ---- stats + per-point max/min over k ----------------------------
        // C/D layout: lane holds N=col, rows M = r + 8*hi; lanes L and L^16
        // carry the same o column -> combine with shfl_xor(16).
#pragma unroll
        for (int t = 0; t < 4; ++t) {
            const v8f c = Cc[t];
            accS[t] += c;
            accQ[t] += c * c;
            float mx = c[0], mn = c[0];
#pragma unroll
            for (int r = 1; r < 8; ++r) {
                mx = fmaxf(mx, c[r]);
                mn = fminf(mn, c[r]);
            }
            mx = fmaxf(mx, __shfl_xor(mx, 16, 32));
            mn = fminf(mn, __shfl_xor(mn, 16, 32));
            if (lane < 16) {
                const size_t o = (size_t)pt * O_ + t * 16 + lane;
                hmax[o] = mx;
                hmin[o] = mn;
            }
        }
    }

    // ---- deterministic block reduction of sum / sumsq ---------------------
#pragma unroll
    for (int t = 0; t < 4; ++t) {
        float s = 0.f, q = 0.f;
#pragma unroll
        for (int r = 0; r < 8; ++r) { s += accS[t][r]; q += accQ[t][r]; }
        s += __shfl_xor(s, 16, 32);
        q += __shfl_xor(q, 16, 32);
        if (lane < 16) {
            redS[wv][t * 16 + lane] = s;
            redQ[wv][t * 16 + lane] = q;
        }
    }
    __syncthreads();
    if (tid < 64) {
        float s = 0.f, q = 0.f;
#pragma unroll
        for (int w = 0; w < 8; ++w) { s += redS[w][tid]; q += redQ[w][tid]; }
        partial[(size_t)blockIdx.x * 128 + tid]      = s;
        partial[(size_t)blockIdx.x * 128 + 64 + tid] = q;
    }
}

__global__ void edgeconv_stats_kernel(const float* __restrict__ partial,
                                      const float* __restrict__ gamma,
                                      const float* __restrict__ beta,
                                      float* __restrict__ ss)
{
    const int o = threadIdx.x;   // 64 threads
    float s = 0.f, q = 0.f;
    for (int blk = 0; blk < BLOCKS_; ++blk) {
        s += partial[blk * 128 + o];
        q += partial[blk * 128 + 64 + o];
    }
    const float inv  = 1.0f / (float)(B_ * N_ * K_);
    const float mean = s * inv;
    const float var  = q * inv - mean * mean;
    const float sc   = gamma[o] * rsqrtf(var + EPS_);
    const float sh   = beta[o] - mean * sc;
    ss[o]      = sc;
    ss[64 + o] = sh;
}

__global__ __launch_bounds__(256)
void edgeconv_apply_kernel(const float* __restrict__ ss,
                           const float* __restrict__ hmin,
                           float* __restrict__ out)   // holds hmax on entry
{
    const size_t i = (size_t)blockIdx.x * 256 + threadIdx.x;
    const int    o = (int)(i & 63);
    const float sc = ss[o];
    const float sh = ss[64 + o];
    // affine monotone in h: positive scale -> max survives, negative -> min
    const float v = (sc >= 0.f) ? out[i] : hmin[i];
    out[i] = fmaxf(0.f, fmaf(v, sc, sh));
}

extern "C" void kernel_launch(void* const* d_in, const int* in_sizes, int n_in,
                              void* d_out, int out_size, void* d_ws, size_t ws_size,
                              hipStream_t stream)
{
    const float* x      = (const float*)d_in[0];
    const int*   idx    = (const int*)  d_in[1];
    const float* weight = (const float*)d_in[2];
    const float* gamma  = (const float*)d_in[3];
    const float* beta   = (const float*)d_in[4];
    float* out = (float*)d_out;

    // ws layout: hmin (B*N*O) | partials (BLOCKS_*128) | scale/shift (128)
    float* hmin    = (float*)d_ws;
    float* partial = hmin + (size_t)B_ * N_ * O_;
    float* ss      = partial + (size_t)BLOCKS_ * 128;

    edgeconv_gemm_kernel<<<BLOCKS_, 256, 0, stream>>>(x, idx, weight,
                                                      out, hmin, partial);
    edgeconv_stats_kernel<<<1, 64, 0, stream>>>(partial, gamma, beta, ss);

    const size_t total = (size_t)B_ * N_ * O_;   // 4,194,304
    edgeconv_apply_kernel<<<total / 256, 256, 0, stream>>>(ss, hmin, out);
}
